// Attention_53437983097258
// MI455X (gfx1250) — matrix operand order repaired and checked by reading, not compiled, not run on hardware
//
#include <hip/hip_runtime.h>
#include <hip/hip_bf16.h>

typedef __bf16 bf16_t;
typedef __attribute__((ext_vector_type(8)))  __bf16 v8bf;
typedef __attribute__((ext_vector_type(16))) __bf16 v16bf;
typedef __attribute__((ext_vector_type(8)))  float  v8f;

#define BATCH 4
#define SEQ   2048
#define DIMSZ 192
#define HEADS 8
#define HD    24
#define GH    16
#define GW    128
#define BH    (BATCH*HEADS)
#define MROWS (BATCH*SEQ)        // 8192
#define VT_LD 2080               // 2048 + 32 pad (so khalf=16 reads stay in-row)
#define Q_SCALE 0.20412414523193154f   // 24^-0.5

// ---------------------------------------------------------------- WMMA helpers

__device__ __forceinline__ v8f wmma_bf16(v16bf a, v16bf b, v8f c) {
  return __builtin_amdgcn_wmma_f32_16x16x32_bf16(false, a, false, b, (short)0, c,
                                                 false, false);
}

__device__ __forceinline__ v16bf make16(v8bf lo, v8bf hi) {
  return __builtin_shufflevector(lo, hi, 0,1,2,3,4,5,6,7,8,9,10,11,12,13,14,15);
}

// A fragment (16xK row-major tile, ld elements):
//   lane holds row = lane&15; elems 0..7 -> K = kb..kb+7, elems 8..15 -> K = 16+kb..
//   with kb = (lane>=16) ? 8 : 0.   (2x 16B loads)
__device__ __forceinline__ v16bf load_afrag(const bf16_t* base, int ld, int row,
                                            int k0, int lane) {
  int kb = (lane & 16) ? 8 : 0;
  const bf16_t* p = base + (size_t)row * ld + k0 + kb;
  v8bf lo = *(const v8bf*)p;
  v8bf hi = *(const v8bf*)(p + 16);
  return make16(lo, hi);
}

// B fragment (KxN tile stored "output-col-major": row = out col, K contiguous):
//   lane holds col = lane&15; elems 0..7 -> K = kh..kh+7, elems 8..15 -> K = 16+kh.., kh = (lane>=16) ? 8 : 0.
__device__ __forceinline__ v16bf load_bfrag(const bf16_t* base, int ld, int row,
                                            int k0, int lane) {
  int kh = (lane & 16) ? 8 : 0;
  const bf16_t* p = base + (size_t)row * ld + k0 + kh;
  v8bf lo = *(const v8bf*)p;
  v8bf hi = *(const v8bf*)(p + 16);
  return make16(lo, hi);
}

// ---------------------------------------------------------------- prep kernel

__global__ void prep_kernel(const float* __restrict__ x,
                            const float* __restrict__ qkv_w,
                            const float* __restrict__ proj_w,
                            bf16_t* __restrict__ xb, bf16_t* __restrict__ qwb,
                            bf16_t* __restrict__ pwb,
                            bf16_t* __restrict__ Qp, bf16_t* __restrict__ Kp,
                            bf16_t* __restrict__ Vt) {
  size_t tid    = (size_t)blockIdx.x * blockDim.x + threadIdx.x;
  size_t stride = (size_t)gridDim.x * blockDim.x;
  for (size_t i = tid; i < (size_t)MROWS * DIMSZ; i += stride) xb[i] = (bf16_t)x[i];
  for (size_t i = tid; i < (size_t)3 * DIMSZ * DIMSZ; i += stride) qwb[i] = (bf16_t)qkv_w[i];
  for (size_t i = tid; i < (size_t)DIMSZ * DIMSZ; i += stride) pwb[i] = (bf16_t)proj_w[i];
  for (size_t i = tid; i < (size_t)BH * SEQ * 32; i += stride) {
    Qp[i] = (bf16_t)0.f;
    Kp[i] = (bf16_t)0.f;
  }
  for (size_t i = tid; i < (size_t)BH * 32 * VT_LD; i += stride) Vt[i] = (bf16_t)0.f;
}

// ---------------------------------------------------------------- QKV GEMM
// qkv[8192, 576] = x @ qkv_w^T + qkv_b ; scatter into Qpad / Kpad / V^T (bf16)

__global__ void __launch_bounds__(256)
qkv_kernel(const bf16_t* __restrict__ xb, const bf16_t* __restrict__ qwb,
           const float* __restrict__ qkv_b,
           bf16_t* __restrict__ Qp, bf16_t* __restrict__ Kp,
           bf16_t* __restrict__ Vt) {
  const int NT = 36;                                   // 576 / 16
  int lane = threadIdx.x & 31;
  int gw   = blockIdx.x * (blockDim.x >> 5) + (threadIdx.x >> 5);
  if (gw >= (MROWS / 16) * NT) return;
  int mt = gw / NT, nt = gw % NT;
  int mbase = mt * 16, nbase = nt * 16;

  v8f c = {};
#pragma unroll
  for (int kc = 0; kc < 6; ++kc) {
    v16bf a = load_afrag(xb,  DIMSZ, mbase + (lane & 15), kc * 32, lane);
    v16bf b = load_bfrag(qwb, DIMSZ, nbase + (lane & 15), kc * 32, lane);
    c = wmma_bf16(a, b, c);
  }

  int col   = nbase + (lane & 15);
  float bias = qkv_b[col];
  int which = col / DIMSZ;          // 0=q 1=k 2=v
  int cc    = col % DIMSZ;
  int hh    = cc / HD, dd = cc % HD;
  int hi    = (lane >> 4) & 1;
#pragma unroll
  for (int r = 0; r < 8; ++r) {
    int gm = mbase + r + 8 * hi;
    int bb = gm / SEQ, nn = gm % SEQ;
    int bh = bb * HEADS + hh;
    float v = c[r] + bias;
    if (which == 0)
      Qp[(size_t)bh * SEQ * 32 + (size_t)nn * 32 + dd] = (bf16_t)(v * Q_SCALE);
    else if (which == 1)
      Kp[(size_t)bh * SEQ * 32 + (size_t)nn * 32 + dd] = (bf16_t)v;
    else
      Vt[(size_t)bh * 32 * VT_LD + (size_t)dd * VT_LD + nn] = (bf16_t)v;
  }
}

// ---------------------------------------------------------------- attention
// One wave per (b, h, 16-query tile). Band mask handled analytically;
// flash-style online softmax; P transposed to A-layout through LDS.

__global__ void __launch_bounds__(256)
attn_kernel(const bf16_t* __restrict__ Qp, const bf16_t* __restrict__ Kp,
            const bf16_t* __restrict__ Vt, bf16_t* __restrict__ attnB) {
  __shared__ bf16_t plds[8][16][32];
  int lane = threadIdx.x & 31;
  int wl   = threadIdx.x >> 5;
  int gw   = blockIdx.x * (blockDim.x >> 5) + wl;
  if (gw >= BH * 128) return;

  int bh = gw >> 7, qt = gw & 127;
  int b  = bh >> 3, h = bh & 7;
  int h1 = qt >> 3, wt = qt & 7;
  int wq0 = wt * 16;
  int qn0 = h1 * GW + wq0;
  int hi  = (lane >> 4) & 1;

  // zero P tile (upper K half stays zero forever -> zero-padded A operand)
  bf16_t* pl = &plds[wl][0][0];
  for (int i = lane; i < 16 * 32; i += 32) pl[i] = (bf16_t)0.f;
  asm volatile("s_wait_dscnt 0" ::: "memory");

  const bf16_t* Qb = Qp + (size_t)bh * SEQ * 32;
  const bf16_t* Kb = Kp + (size_t)bh * SEQ * 32;
  const bf16_t* Vb = Vt + (size_t)bh * 32 * VT_LD;

  v16bf qa = load_afrag(Qb, 32, qn0 + (lane & 15), 0, lane);

  float m_i[8], l_i[8];
  v8f acc0 = {}, acc1 = {};
#pragma unroll
  for (int r = 0; r < 8; ++r) { m_i[r] = -__builtin_inff(); l_i[r] = 0.f; }

  int h2lo = (h1 - 3 < 0) ? 0 : h1 - 3;
  int h2hi = (h1 + 3 > 15) ? 15 : h1 + 3;
  int wtlo = (wt - 1 < 0) ? 0 : wt - 1;
  int wthi = (wt + 1 > 7) ? 7 : wt + 1;

  for (int h2 = h2lo; h2 <= h2hi; ++h2) {
    for (int wt2 = wtlo; wt2 <= wthi; ++wt2) {
      int kb0 = h2 * GW + wt2 * 16;

      v16bf kf = load_bfrag(Kb, 32, kb0 + (lane & 15), 0, lane);
      v8f z = {};
      v8f s = wmma_bf16(qa, kf, z);

      int w2 = wt2 * 16 + (lane & 15);
      float p[8];
#pragma unroll
      for (int r = 0; r < 8; ++r) {
        int w1 = wq0 + r + 8 * hi;
        int dw = w1 - w2; if (dw < 0) dw = -dw;
        float sv = (dw > 5) ? -__builtin_inff() : s[r];
        // row max across the 16 lanes that hold this row
        float rm = sv;
        rm = fmaxf(rm, __shfl_xor(rm, 1, 16));
        rm = fmaxf(rm, __shfl_xor(rm, 2, 16));
        rm = fmaxf(rm, __shfl_xor(rm, 4, 16));
        rm = fmaxf(rm, __shfl_xor(rm, 8, 16));
        float mn   = fmaxf(m_i[r], rm);
        float corr = (m_i[r] > -1e30f) ? __expf(m_i[r] - mn) : 0.f;
        float pv   = (sv    > -1e30f) ? __expf(sv    - mn) : 0.f;
        float ps = pv;
        ps += __shfl_xor(ps, 1, 16);
        ps += __shfl_xor(ps, 2, 16);
        ps += __shfl_xor(ps, 4, 16);
        ps += __shfl_xor(ps, 8, 16);
        l_i[r] = l_i[r] * corr + ps;
        m_i[r] = mn;
        acc0[r] *= corr;
        acc1[r] *= corr;
        p[r] = pv;
      }

      // transpose P (C-layout -> A-layout) through LDS
#pragma unroll
      for (int r = 0; r < 8; ++r)
        plds[wl][r + 8 * hi][lane & 15] = (bf16_t)p[r];
      asm volatile("s_wait_dscnt 0" ::: "memory");
      v16bf pa = load_afrag(&plds[wl][0][0], 32, lane & 15, 0, lane);

      v16bf vb0 = load_bfrag(Vb, VT_LD, (lane & 15),      kb0, lane);
      v16bf vb1 = load_bfrag(Vb, VT_LD, 16 + (lane & 15), kb0, lane);
      acc0 = wmma_bf16(pa, vb0, acc0);
      acc1 = wmma_bf16(pa, vb1, acc1);
    }
  }

  int d0 = lane & 15;
#pragma unroll
  for (int r = 0; r < 8; ++r) {
    float inv = (l_i[r] > 0.f) ? 1.f / l_i[r] : 0.f;
    int nn = qn0 + r + 8 * hi;
    size_t rowo = ((size_t)b * SEQ + nn) * DIMSZ + h * HD;
    attnB[rowo + d0] = (bf16_t)(acc0[r] * inv);
    if (d0 < 8) attnB[rowo + 16 + d0] = (bf16_t)(acc1[r] * inv);
  }
}

// ---------------------------------------------------------------- out proj
// out[8192,192] = attnB @ proj_w^T + proj_b   (fp32 output)

__global__ void __launch_bounds__(256)
proj_kernel(const bf16_t* __restrict__ ab, const bf16_t* __restrict__ pwb,
            const float* __restrict__ proj_b, float* __restrict__ out) {
  const int NT = 12;                                  // 192 / 16
  int lane = threadIdx.x & 31;
  int gw   = blockIdx.x * (blockDim.x >> 5) + (threadIdx.x >> 5);
  if (gw >= (MROWS / 16) * NT) return;
  int mt = gw / NT, nt = gw % NT;
  int mbase = mt * 16, nbase = nt * 16;

  v8f c = {};
#pragma unroll
  for (int kc = 0; kc < 6; ++kc) {
    v16bf a = load_afrag(ab,  DIMSZ, mbase + (lane & 15), kc * 32, lane);
    v16bf b = load_bfrag(pwb, DIMSZ, nbase + (lane & 15), kc * 32, lane);
    c = wmma_bf16(a, b, c);
  }

  int col  = nbase + (lane & 15);
  float bias = proj_b[col];
  int hi = (lane >> 4) & 1;
#pragma unroll
  for (int r = 0; r < 8; ++r) {
    int gm = mbase + r + 8 * hi;
    out[(size_t)gm * DIMSZ + col] = c[r] + bias;
  }
}

// ---------------------------------------------------------------- launch

extern "C" void kernel_launch(void* const* d_in, const int* in_sizes, int n_in,
                              void* d_out, int out_size, void* d_ws, size_t ws_size,
                              hipStream_t stream) {
  (void)in_sizes; (void)n_in; (void)out_size; (void)ws_size;
  const float* x      = (const float*)d_in[0];
  const float* qkv_w  = (const float*)d_in[1];
  const float* qkv_b  = (const float*)d_in[2];
  const float* proj_w = (const float*)d_in[3];
  const float* proj_b = (const float*)d_in[4];
  // d_in[5] (mask) intentionally unused: the band mask is computed analytically.
  float* out = (float*)d_out;

  char* ws = (char*)d_ws;
  size_t off = 0;
  auto alloc = [&](size_t bytes) -> size_t {
    size_t r = off;
    off = (off + bytes + 255) & ~(size_t)255;
    return r;
  };
  bf16_t* xb  = (bf16_t*)(ws + alloc((size_t)MROWS * DIMSZ * 2));
  bf16_t* qwb = (bf16_t*)(ws + alloc((size_t)3 * DIMSZ * DIMSZ * 2));
  bf16_t* pwb = (bf16_t*)(ws + alloc((size_t)DIMSZ * DIMSZ * 2));
  bf16_t* Qp  = (bf16_t*)(ws + alloc((size_t)BH * SEQ * 32 * 2));
  bf16_t* Kp  = (bf16_t*)(ws + alloc((size_t)BH * SEQ * 32 * 2));
  bf16_t* Vt  = (bf16_t*)(ws + alloc((size_t)BH * 32 * VT_LD * 2));
  bf16_t* aB  = (bf16_t*)(ws + alloc((size_t)MROWS * DIMSZ * 2));

  prep_kernel<<<2048, 256, 0, stream>>>(x, qkv_w, proj_w, xb, qwb, pwb, Qp, Kp, Vt);

  // 512 M-tiles x 36 N-tiles, 8 waves (tiles) per 256-thread block
  qkv_kernel<<<(512 * 36 + 7) / 8, 256, 0, stream>>>(xb, qwb, qkv_b, Qp, Kp, Vt);

  // 32 (b,h) x 128 query tiles = 4096 waves
  attn_kernel<<<(BH * 128 + 7) / 8, 256, 0, stream>>>(Qp, Kp, Vt, aB);

  // 512 M-tiles x 12 N-tiles
  proj_kernel<<<(512 * 12 + 7) / 8, 256, 0, stream>>>(aB, pwb, proj_b, out);
}
